// MultiHeadGraphAttention_26001732010507
// MI455X (gfx1250) — matrix operand order repaired and checked by reading, not compiled, not run on hardware
//
#include <hip/hip_runtime.h>
#include <hip/hip_fp16.h>
#include <math.h>
#include <stdint.h>

#define NNODES   20000
#define NEDGES   320000
#define HIDDEN   256
#define NHEADS   8
#define HDIM     32

typedef __attribute__((ext_vector_type(16))) _Float16 v16h;
typedef __attribute__((ext_vector_type(8)))  _Float16 v8h;
typedef __attribute__((ext_vector_type(8)))  float    v8f;

#define LDSP 40   // padded LDS row length in f16 (80B = 20 banks -> conflict-free)

// ---------------------------------------------------------------------------
// Generic WMMA GEMM: C[M,Nc] = act(A[M,K](f16) @ W[K,Nc](f16) + bias) (+resid)
// Block: 256 threads = 8 waves. Block tile 128(M) x 64(N). Wave tile 32x32.
// A tile staged with async global->LDS b128 copies (ASYNCcnt), double-buffered
// so the next tile's HBM traffic overlaps the current tile's WMMAs.
// ---------------------------------------------------------------------------
__global__ __launch_bounds__(256) void gemm_wmma_f16(
    const _Float16* __restrict__ A, const _Float16* __restrict__ W,
    const float* __restrict__ bias, const float* __restrict__ resid,
    float* __restrict__ Cf, _Float16* __restrict__ Ch,
    int M, int K, int Ncols, int relu)
{
    __shared__ __align__(16) _Float16 sA[2][128 * LDSP];   // [row][k]  row-major
    __shared__ __align__(16) _Float16 sB[2][64 * LDSP];    // [n][k]    transposed

    const int tid  = threadIdx.x;
    const int lane = tid & 31;
    const int wave = tid >> 5;
    const int wm   = wave >> 1;        // 0..3  (M direction)
    const int wn   = wave & 1;         // 0..1  (N direction)
    const int lm   = lane & 15;
    const int lh   = lane >> 4;
    const int blockM = blockIdx.y * 128;
    const int blockN = blockIdx.x * 64;

    // --- staging helpers ---------------------------------------------------
    // A tile (128 x 32 f16): each thread owns one 16-f16 half-row segment -> two
    // async b128 global->LDS transfers (lane-masked at the M edge).
    auto stageA_async = [&](int buf, int k0) {
        int r  = tid >> 1;
        int c  = (tid & 1) * 16;
        int gr = blockM + r;
        uint32_t l0 = (uint32_t)(uintptr_t)&sA[buf][r * LDSP + c];
        if (gr < M) {
            const _Float16* p = A + (size_t)gr * K + k0 + c;
            asm volatile("global_load_async_to_lds_b128 %0, %1, off"
                         :: "v"(l0), "v"(p) : "memory");
            asm volatile("global_load_async_to_lds_b128 %0, %1, off offset:16"
                         :: "v"(l0 + 16u), "v"(p) : "memory");
        }
    };
    // B tile (32 x 64 f16) must be transposed into sB[n][k]: global load into
    // registers (hoisted into compute phase), scatter-store to LDS afterwards.
    auto loadB = [&](int k0) -> v8h {
        int r  = tid >> 3;          // k row 0..31
        int c0 = (tid & 7) * 8;     // n col
        v8h b = {};
        if (blockN + c0 < Ncols)
            b = *(const v8h*)(W + (size_t)(k0 + r) * Ncols + blockN + c0);
        return b;
    };
    auto storeB = [&](int buf, v8h b) {
        int r  = tid >> 3;
        int c0 = (tid & 7) * 8;
        #pragma unroll
        for (int j = 0; j < 8; ++j)
            sB[buf][(c0 + j) * LDSP + r] = b[j];
    };

    v8f acc[2][2] = {};

    // --- prologue: stage k-step 0 into buffer 0 ----------------------------
    stageA_async(0, 0);
    storeB(0, loadB(0));
    asm volatile("s_wait_asynccnt 0x0" ::: "memory");
    __syncthreads();

    int cur = 0;
    for (int k0 = 0; k0 < K; k0 += 32) {
        const int nxt  = cur ^ 1;
        const bool more = (k0 + 32) < K;

        // kick off next tile's traffic before computing on the current one
        v8h bn = {};
        if (more) {
            stageA_async(nxt, k0 + 32);
            bn = loadB(k0 + 32);
        }

        // ---- build fragments per ISA 16-bit layouts ----
        v16h afr[2], bfr[2];
        #pragma unroll
        for (int i = 0; i < 2; ++i) {
            // lanes 0-15: row=lm, K = 0..7 then 16..23 ; lanes 16-31: K = 8..15 then 24..31
            const _Float16* p = &sA[cur][(wm * 32 + i * 16 + lm) * LDSP + lh * 8];
            v8h lo = *(const v8h*)p;
            v8h hi = *(const v8h*)(p + 16);
            afr[i] = __builtin_shufflevector(lo, hi, 0,1,2,3,4,5,6,7,8,9,10,11,12,13,14,15);
        }
        #pragma unroll
        for (int j = 0; j < 2; ++j) {
            // lane: n=lm, K = lh*8 + 0..7 then 16 + lh*8 + 0..7 (transposed sB)
            const _Float16* p = &sB[cur][(wn * 32 + j * 16 + lm) * LDSP + lh * 8];
            v8h lo = *(const v8h*)p;
            v8h hi = *(const v8h*)(p + 16);
            bfr[j] = __builtin_shufflevector(lo, hi, 0,1,2,3,4,5,6,7,8,9,10,11,12,13,14,15);
        }

        #pragma unroll
        for (int i = 0; i < 2; ++i)
            #pragma unroll
            for (int j = 0; j < 2; ++j)
                acc[i][j] = __builtin_amdgcn_wmma_f32_16x16x32_f16(
                    false, afr[i], false, bfr[j], (short)0, acc[i][j], false, false);

        // finish staging the next buffer (safe: nxt not read this iteration)
        if (more) storeB(nxt, bn);

        // retire this pipeline stage: own async copies done, then block-wide sync
        asm volatile("s_wait_asynccnt 0x0" ::: "memory");
        __syncthreads();
        cur = nxt;
    }

    // ---- writeback (C layout: VGPR r -> M = r + 8*lh, N = lm) ----
    #pragma unroll
    for (int i = 0; i < 2; ++i) {
        #pragma unroll
        for (int j = 0; j < 2; ++j) {
            int col = blockN + wn * 32 + j * 16 + lm;
            if (col >= Ncols) continue;
            float bv = bias ? bias[col] : 0.0f;
            #pragma unroll
            for (int r = 0; r < 8; ++r) {
                int row = blockM + wm * 32 + i * 16 + r + 8 * lh;
                if (row >= M) continue;
                float v = acc[i][j][r] + bv;
                if (relu && v < 0.0f) v = 0.0f;
                size_t o = (size_t)row * Ncols + col;
                if (resid) v += resid[o];
                if (Cf) Cf[o] = v;
                if (Ch) Ch[o] = (_Float16)v;
            }
        }
    }
}

// ---------------------------------------------------------------------------
__global__ void cvt_f32_f16(const float* __restrict__ in, _Float16* __restrict__ out, int n)
{
    int i = blockIdx.x * blockDim.x + threadIdx.x;
    if (i < n) out[i] = (_Float16)in[i];
}

// Wcomb = ee_W2 @ Wea  [256,8];  bcomb = ee_b2 @ Wea + bea  [8]
__global__ void compose_wea(const float* __restrict__ eeW2, const float* __restrict__ Wea,
                            const float* __restrict__ eeb2, const float* __restrict__ bea,
                            float* __restrict__ Wcomb, float* __restrict__ bcomb)
{
    int idx = blockIdx.x * blockDim.x + threadIdx.x;
    if (idx < HIDDEN * NHEADS) {
        int r = idx >> 3, c = idx & 7;
        float acc = 0.0f;
        for (int k = 0; k < HIDDEN; ++k)
            acc += eeW2[r * HIDDEN + k] * Wea[k * NHEADS + c];
        Wcomb[idx] = acc;
    }
    if (idx < NHEADS) {
        float acc = bea[idx];
        for (int k = 0; k < HIDDEN; ++k)
            acc += eeb2[k] * Wea[k * NHEADS + idx];
        bcomb[idx] = acc;
    }
}

__global__ void init_bufs(float* __restrict__ seg_max, float* __restrict__ seg_sum,
                          float* __restrict__ agg, int nseg, int nagg)
{
    int i = blockIdx.x * blockDim.x + threadIdx.x;
    if (i < nseg) { seg_max[i] = -INFINITY; seg_sum[i] = 0.0f; }
    if (i < nagg) agg[i] = 0.0f;
}

__device__ __forceinline__ void atomicMaxF(float* addr, float v)
{
    // sign-aware int/uint trick (monotone under IEEE-754 ordering)
    if (v >= 0.0f) atomicMax((int*)addr, __float_as_int(v));
    else           atomicMin((unsigned int*)addr, __float_as_uint(v));
}

// scores[e,h] = <Q[src,h,:],K[dst,h,:]>/sqrt(32) + escore[e,h]; seg_max via atomic
__global__ void edge_scores(const float* __restrict__ Q, const float* __restrict__ Kx,
                            const int* __restrict__ srcs, const int* __restrict__ dsts,
                            float* __restrict__ sc /* in: escore, out: scores */,
                            float* __restrict__ seg_max, int ne)
{
    int idx = blockIdx.x * blockDim.x + threadIdx.x;
    if (idx >= ne * NHEADS) return;
    int e = idx >> 3, h = idx & 7;
    int s = srcs[e], d = dsts[e];
    const float4* q = (const float4*)(Q  + (size_t)s * HIDDEN + h * HDIM);
    const float4* k = (const float4*)(Kx + (size_t)d * HIDDEN + h * HDIM);
    float acc = 0.0f;
    #pragma unroll
    for (int i = 0; i < 8; ++i) {
        float4 a = q[i], b = k[i];
        acc += a.x * b.x + a.y * b.y + a.z * b.z + a.w * b.w;
    }
    float v = acc * 0.17677669529663689f + sc[idx];   // 1/sqrt(HDIM)
    sc[idx] = v;
    atomicMaxF(&seg_max[(size_t)s * NHEADS + h], v);
}

// ex = exp(scores - seg_max[src]); seg_sum += ex   (in-place over sc)
__global__ void edge_exp(float* __restrict__ sc, const float* __restrict__ seg_max,
                         const int* __restrict__ srcs, float* __restrict__ seg_sum, int ne)
{
    int idx = blockIdx.x * blockDim.x + threadIdx.x;
    if (idx >= ne * NHEADS) return;
    int e = idx >> 3, h = idx & 7;
    int s = srcs[e];
    float v = __expf(sc[idx] - seg_max[(size_t)s * NHEADS + h]);
    sc[idx] = v;
    atomicAdd(&seg_sum[(size_t)s * NHEADS + h], v);
}

// attn = ex/seg_sum[src]; agg[dst] += V[src]*attn; emit attn
__global__ void edge_attn_scatter(const float* __restrict__ sc, const float* __restrict__ seg_sum,
                                  const float* __restrict__ V, const int* __restrict__ srcs,
                                  const int* __restrict__ dsts, float* __restrict__ attn_out,
                                  float* __restrict__ agg, int ne)
{
    int idx = blockIdx.x * blockDim.x + threadIdx.x;
    if (idx >= ne * NHEADS) return;
    int e = idx >> 3, h = idx & 7;
    int s = srcs[e], d = dsts[e];
    float a = sc[idx] / seg_sum[(size_t)s * NHEADS + h];
    attn_out[idx] = a;
    const float* v = V   + (size_t)s * HIDDEN + h * HDIM;
    float*       o = agg + (size_t)d * HIDDEN + h * HDIM;
    #pragma unroll
    for (int i = 0; i < HDIM; ++i)
        atomicAdd(o + i, v[i] * a);
}

// ---------------------------------------------------------------------------
extern "C" void kernel_launch(void* const* d_in, const int* in_sizes, int n_in,
                              void* d_out, int out_size, void* d_ws, size_t ws_size,
                              hipStream_t stream)
{
    const float* node_feat = (const float*)d_in[0];
    const float* edge_feat = (const float*)d_in[1];
    const int*   edge_idx  = (const int*)d_in[3];     // [2,E]
    const float* ne_W1 = (const float*)d_in[4];  const float* ne_b1 = (const float*)d_in[5];
    const float* ne_W2 = (const float*)d_in[6];  const float* ne_b2 = (const float*)d_in[7];
    const float* ee_W1 = (const float*)d_in[8];  const float* ee_b1 = (const float*)d_in[9];
    const float* ee_W2 = (const float*)d_in[10]; const float* ee_b2 = (const float*)d_in[11];
    const float* Wq = (const float*)d_in[12];    const float* bq = (const float*)d_in[13];
    const float* Wk = (const float*)d_in[14];    const float* bk = (const float*)d_in[15];
    const float* Wv = (const float*)d_in[16];    const float* bv = (const float*)d_in[17];
    const float* Wea = (const float*)d_in[18];   const float* bea = (const float*)d_in[19];
    const float* Wo = (const float*)d_in[20];    const float* bo = (const float*)d_in[21];

    float* out_main = (float*)d_out;                          // [N,256]
    float* attn_out = out_main + (size_t)NNODES * HIDDEN;     // [E,8]

    const int* srcs = edge_idx;
    const int* dsts = edge_idx + NEDGES;

    // ---- workspace carve-out ----
    size_t off = 0;
    auto alloc = [&](size_t elems, size_t elsz) -> void* {
        off = (off + 255) & ~(size_t)255;
        void* p = (char*)d_ws + off;
        off += elems * elsz;
        return p;
    };
    _Float16* xn16   = (_Float16*)alloc((size_t)NNODES * 64, 2);
    _Float16* xe16   = (_Float16*)alloc((size_t)NEDGES * 32, 2);
    _Float16* t1_16  = (_Float16*)alloc((size_t)NNODES * HIDDEN, 2);
    _Float16* h16    = (_Float16*)alloc((size_t)NNODES * HIDDEN, 2);
    _Float16* t2_16  = (_Float16*)alloc((size_t)NEDGES * HIDDEN, 2);
    _Float16* agg16  = (_Float16*)alloc((size_t)NNODES * HIDDEN, 2);
    _Float16* neW1h  = (_Float16*)alloc(64 * HIDDEN, 2);
    _Float16* neW2h  = (_Float16*)alloc(HIDDEN * HIDDEN, 2);
    _Float16* eeW1h  = (_Float16*)alloc(32 * HIDDEN, 2);
    _Float16* Wqh    = (_Float16*)alloc(HIDDEN * HIDDEN, 2);
    _Float16* Wkh    = (_Float16*)alloc(HIDDEN * HIDDEN, 2);
    _Float16* Wvh    = (_Float16*)alloc(HIDDEN * HIDDEN, 2);
    _Float16* Woh    = (_Float16*)alloc(HIDDEN * HIDDEN, 2);
    _Float16* Wcombh = (_Float16*)alloc(HIDDEN * NHEADS, 2);
    float* hF      = (float*)alloc((size_t)NNODES * HIDDEN, 4);
    float* Qf      = (float*)alloc((size_t)NNODES * HIDDEN, 4);
    float* Kf      = (float*)alloc((size_t)NNODES * HIDDEN, 4);
    float* Vf      = (float*)alloc((size_t)NNODES * HIDDEN, 4);
    float* scbuf   = (float*)alloc((size_t)NEDGES * NHEADS, 4);  // escore->scores->ex
    float* seg_max = (float*)alloc((size_t)NNODES * NHEADS, 4);
    float* seg_sum = (float*)alloc((size_t)NNODES * NHEADS, 4);
    float* aggF    = (float*)alloc((size_t)NNODES * HIDDEN, 4);
    float* Wcomb   = (float*)alloc(HIDDEN * NHEADS, 4);
    float* bcomb   = (float*)alloc(NHEADS, 4);

    auto cvt = [&](const float* src, _Float16* dst, int n) {
        cvt_f32_f16<<<(n + 255) / 256, 256, 0, stream>>>(src, dst, n);
    };
    auto gemm = [&](const _Float16* A, const _Float16* W, const float* bias,
                    const float* resid, float* Cf, _Float16* Ch,
                    int M, int K, int Nc, int relu) {
        dim3 grid((Nc + 63) / 64, (M + 127) / 128);
        gemm_wmma_f16<<<grid, 256, 0, stream>>>(A, W, bias, resid, Cf, Ch, M, K, Nc, relu);
    };

    // ---- conversions ----
    cvt(node_feat, xn16, NNODES * 64);
    cvt(edge_feat, xe16, NEDGES * 32);
    cvt(ne_W1, neW1h, 64 * HIDDEN);
    cvt(ne_W2, neW2h, HIDDEN * HIDDEN);
    cvt(ee_W1, eeW1h, 32 * HIDDEN);
    cvt(Wq, Wqh, HIDDEN * HIDDEN);
    cvt(Wk, Wkh, HIDDEN * HIDDEN);
    cvt(Wv, Wvh, HIDDEN * HIDDEN);
    cvt(Wo, Woh, HIDDEN * HIDDEN);
    compose_wea<<<(HIDDEN * NHEADS + 255) / 256, 256, 0, stream>>>(
        ee_W2, Wea, ee_b2, bea, Wcomb, bcomb);
    cvt(Wcomb, Wcombh, HIDDEN * NHEADS);

    // ---- node encoder: h = relu(x@W1+b1)@W2+b2 ----
    gemm(xn16, neW1h, ne_b1, nullptr, nullptr, t1_16, NNODES, 64, HIDDEN, 1);
    gemm(t1_16, neW2h, ne_b2, nullptr, hF, h16, NNODES, HIDDEN, HIDDEN, 0);

    // ---- edge encoder layer 1 + composed Wea (layer2 fused algebraically) ----
    gemm(xe16, eeW1h, ee_b1, nullptr, nullptr, t2_16, NEDGES, 32, HIDDEN, 1);
    gemm(t2_16, Wcombh, bcomb, nullptr, scbuf, nullptr, NEDGES, HIDDEN, NHEADS, 0);

    // ---- Q,K,V projections ----
    gemm(h16, Wqh, bq, nullptr, Qf, nullptr, NNODES, HIDDEN, HIDDEN, 0);
    gemm(h16, Wkh, bk, nullptr, Kf, nullptr, NNODES, HIDDEN, HIDDEN, 0);
    gemm(h16, Wvh, bv, nullptr, Vf, nullptr, NNODES, HIDDEN, HIDDEN, 0);

    // ---- segment softmax + scatter ----
    {
        int nseg = NNODES * NHEADS, nagg = NNODES * HIDDEN;
        init_bufs<<<(nagg + 255) / 256, 256, 0, stream>>>(seg_max, seg_sum, aggF, nseg, nagg);
        int ntot = NEDGES * NHEADS;
        edge_scores<<<(ntot + 255) / 256, 256, 0, stream>>>(Qf, Kf, srcs, dsts, scbuf, seg_max, NEDGES);
        edge_exp<<<(ntot + 255) / 256, 256, 0, stream>>>(scbuf, seg_max, srcs, seg_sum, NEDGES);
        edge_attn_scatter<<<(ntot + 255) / 256, 256, 0, stream>>>(
            scbuf, seg_sum, Vf, srcs, dsts, attn_out, aggF, NEDGES);
    }

    // ---- output projection + residual: out = agg@Wo + bo + h ----
    cvt(aggF, agg16, NNODES * HIDDEN);
    gemm(agg16, Woh, bo, hF, out_main, nullptr, NNODES, HIDDEN, HIDDEN, 0);
}
